// MambaBlock_44461501448810
// MI455X (gfx1250) — matrix operand order repaired and checked by reading, not compiled, not run on hardware
//
#include <hip/hip_runtime.h>
#include <hip/hip_bf16.h>
#include <math.h>

// ---- problem dims (match reference) ----
#define DM   256          // D_MODEL
#define DI   1024         // D_INNER
#define DS   16           // D_STATE
#define DC   4            // D_CONV
#define DTR  16           // DT_RANK
#define NB   2            // BATCH
#define LSEQ 2048         // SEQ
#define NROW (NB*LSEQ)    // 4096 rows per direction
#define XPN  (DTR + 2*DS) // 48

typedef __bf16 bf16_t;
typedef __attribute__((ext_vector_type(16))) __bf16 v16bf;
typedef __attribute__((ext_vector_type(8)))  __bf16 v8bf;
typedef __attribute__((ext_vector_type(8)))  float  v8f;

__device__ __forceinline__ float siluf(float v) { return v / (1.0f + __expf(-v)); }

// low 32 bits of a flat shared-aperture address == wave-relative LDS byte offset
__device__ __forceinline__ unsigned lds_off(const void* p) {
  return (unsigned)(size_t)p;
}

// CDNA5 async copy: 16B global -> LDS, tracked by ASYNCcnt (ISA 10.x / 15.18)
__device__ __forceinline__ void async_ld16(unsigned lds, const float* gp) {
  asm volatile("global_load_async_to_lds_b128 %0, %1, off"
               :: "v"(lds), "v"((unsigned long long)(size_t)gp)
               : "memory");
}

// ---------------- f32 -> bf16 pack ----------------
__global__ void k_pack(const float* __restrict__ src, bf16_t* __restrict__ dst, int n) {
  int i = blockIdx.x * blockDim.x + threadIdx.x;
  if (i < n) dst[i] = (bf16_t)src[i];
}

// ---------------- RMSNorm (optionally sequence-flipped read) ----------------
__global__ __launch_bounds__(DM) void k_rmsnorm(const float* __restrict__ x,
                                                const float* __restrict__ w,
                                                bf16_t* __restrict__ out, int flip) {
  __shared__ float red[DM];
  int m = blockIdx.x;            // 0..NROW-1 (destination row, direction-local order)
  int b = m / LSEQ, l = m % LSEQ;
  int sl = flip ? (LSEQ - 1 - l) : l;
  int d = threadIdx.x;
  float v = x[((size_t)b * LSEQ + sl) * DM + d];
  red[d] = v * v;
  __syncthreads();
  for (int s = DM / 2; s > 0; s >>= 1) {
    if (d < s) red[d] += red[d + s];
    __syncthreads();
  }
  float scale = rsqrtf(red[0] * (1.0f / DM) + 1e-5f);
  out[(size_t)m * DM + d] = (bf16_t)(v * scale * w[d]);
}

// ---------------- WMMA bf16 GEMM: C[M,N] = A[M,K] * W[N,K]^T (+bias) ----------------
// One wave computes a 32x16 tile: two v_wmma_f32_16x16x32_bf16 sharing one B fragment.
__global__ __launch_bounds__(32) void k_gemm(const bf16_t* __restrict__ A,
                                             const bf16_t* __restrict__ W,
                                             float* __restrict__ C,
                                             const float* __restrict__ bias,
                                             int M, int N, int K) {
  int n0 = blockIdx.x * 16;
  int m0 = blockIdx.y * 32;
  int lane = threadIdx.x;
  int half = lane >> 4;          // 0: lanes 0-15, 1: lanes 16-31
  int idx  = lane & 15;
  // A fragment: lane holds row; half 0 -> K {0..7,16..23}, half 1 -> K {8..15,24..31}
  const bf16_t* ap0 = A + (size_t)(m0 + idx) * K + (half ? 8 : 0);
  const bf16_t* ap1 = ap0 + (size_t)16 * K;
  // B fragment (B = W^T, 32x16): lane holds column; half 0 -> K {0..7,16..23}, half 1 -> K {8..15,24..31}
  const bf16_t* wp = W + (size_t)(n0 + idx) * K + (half ? 8 : 0);
  v8f acc0 = {0.f, 0.f, 0.f, 0.f, 0.f, 0.f, 0.f, 0.f};
  v8f acc1 = {0.f, 0.f, 0.f, 0.f, 0.f, 0.f, 0.f, 0.f};
  for (int k = 0; k < K; k += 32) {
    v8bf blo = *(const v8bf*)(wp + k);
    v8bf bhi = *(const v8bf*)(wp + k + 16);
    v8bf a0l = *(const v8bf*)(ap0 + k);
    v8bf a0h = *(const v8bf*)(ap0 + k + 16);
    v8bf a1l = *(const v8bf*)(ap1 + k);
    v8bf a1h = *(const v8bf*)(ap1 + k + 16);
    v16bf bv = __builtin_shufflevector(blo, bhi, 0,1,2,3,4,5,6,7,8,9,10,11,12,13,14,15);
    v16bf a0 = __builtin_shufflevector(a0l, a0h, 0,1,2,3,4,5,6,7,8,9,10,11,12,13,14,15);
    v16bf a1 = __builtin_shufflevector(a1l, a1h, 0,1,2,3,4,5,6,7,8,9,10,11,12,13,14,15);
    acc0 = __builtin_amdgcn_wmma_f32_16x16x32_bf16(false, a0, false, bv,
                                                   (short)0, acc0, false, false);
    acc1 = __builtin_amdgcn_wmma_f32_16x16x32_bf16(false, a1, false, bv,
                                                   (short)0, acc1, false, false);
  }
  float bv = bias ? bias[n0 + idx] : 0.0f;
  int row = m0 + 8 * half;       // C/D layout: VGPR r -> M = r + 8*half, lane -> N = idx
  #pragma unroll
  for (int r = 0; r < 8; ++r) {
    C[(size_t)(row + r) * N + (n0 + idx)] = acc0[r] + bv;
    C[(size_t)(row + 16 + r) * N + (n0 + idx)] = acc1[r] + bv;
  }
}

// ---------------- causal depthwise conv (k=4) + bias + SiLU ----------------
__global__ void k_conv(const float* __restrict__ xz, const float* __restrict__ cw,
                       const float* __restrict__ cb, float* __restrict__ xs,
                       bf16_t* __restrict__ xsb) {
  int i = blockIdx.x * blockDim.x + threadIdx.x;   // over NROW*DI
  if (i >= NROW * DI) return;
  int ch = i % DI, m = i / DI;
  int b = m / LSEQ, l = m % LSEQ;
  float s = cb[ch];
  #pragma unroll
  for (int j = 0; j < DC; ++j) {
    int ll = l - (DC - 1) + j;
    if (ll >= 0)
      s += xz[((size_t)(b * LSEQ + ll)) * (2 * DI) + ch] * cw[ch * DC + j];
  }
  float v = siluf(s);
  xs[i] = v;
  xsb[i] = (bf16_t)v;
}

// ---------------- dt = softplus(dt_low @ dt_w^T + dt_b);  u = dt * xs ----------------
__global__ void k_dtu(const float* __restrict__ dbl, const float* __restrict__ dtw,
                      const float* __restrict__ dtb, const float* __restrict__ xs,
                      float* __restrict__ dt, float* __restrict__ u) {
  int i = blockIdx.x * blockDim.x + threadIdx.x;
  if (i >= NROW * DI) return;
  int ch = i % DI, m = i / DI;
  const float* dr = dbl + (size_t)m * XPN;
  float a = dtb[ch];
  #pragma unroll
  for (int r = 0; r < DTR; ++r) a += dr[r] * dtw[ch * DTR + r];
  float dtv = (a > 20.0f) ? a : log1pf(__expf(a));
  dt[i] = dtv;
  u[i] = dtv * xs[i];
}

// ---------------- sequential selective scan, async double-buffered LDS staging ----
// Block: 256 threads = 16 channels x 16 states, one (batch) sequence slice.
// Per 16-timestep chunk each thread issues exactly one async b128 global->LDS copy
// (branch is uniform per wave), overlapped with consumption of the previous chunk.
#define SCH 16   // channels per block
#define TCH 16   // timesteps per chunk
__global__ __launch_bounds__(256) void k_scan(const float* __restrict__ dt,
                                              const float* __restrict__ u,
                                              const float* __restrict__ dbl,
                                              const float* __restrict__ Alog,
                                              float* __restrict__ ys) {
  __shared__ float dtT[2][TCH][SCH];
  __shared__ float uT [2][TCH][SCH];
  __shared__ float bcT[2][TCH][2 * DS];
  int b   = blockIdx.y;
  int tid = threadIdx.x;
  int ch0 = blockIdx.x * SCH;
  int chl = tid >> 4;
  int st  = tid & 15;
  int ch  = ch0 + chl;
  const float* dtp = dt  + (size_t)b * LSEQ * DI;
  const float* up  = u   + (size_t)b * LSEQ * DI;
  const float* dp  = dbl + (size_t)b * LSEQ * XPN;
  float* yp = ys + (size_t)b * LSEQ * DI;
  float Av = -__expf(Alog[ch * DS + st]);
  float h = 0.0f;

  auto issue = [&](int t0, int buf) {
    if (tid < 64) {          // waves 0-1: dt tile (16 rows x 64B)
      int row = tid >> 2, q = tid & 3;
      async_ld16(lds_off(&dtT[buf][row][q * 4]),
                 dtp + (size_t)(t0 + row) * DI + ch0 + q * 4);
    } else if (tid < 128) {  // waves 2-3: u tile
      int j = tid - 64, row = j >> 2, q = j & 3;
      async_ld16(lds_off(&uT[buf][row][q * 4]),
                 up + (size_t)(t0 + row) * DI + ch0 + q * 4);
    } else {                 // waves 4-7: B|C tile (16 rows x 128B from dbl cols 16..47)
      int j = tid - 128, row = j >> 3, q = j & 7;
      async_ld16(lds_off(&bcT[buf][row][q * 4]),
                 dp + (size_t)(t0 + row) * XPN + DTR + q * 4);
    }
  };

  constexpr int NCHUNK = LSEQ / TCH;   // 128
  issue(0, 0);
  for (int c = 0; c < NCHUNK; ++c) {
    int buf = c & 1;
    if (c + 1 < NCHUNK) {
      issue((c + 1) * TCH, buf ^ 1);
      asm volatile("s_wait_asynccnt 0x1" ::: "memory");  // this chunk's copies done
    } else {
      asm volatile("s_wait_asynccnt 0x0" ::: "memory");
    }
    __syncthreads();                                     // all waves' tiles visible
    int t0 = c * TCH;
    #pragma unroll 4
    for (int i = 0; i < TCH; ++i) {
      float dtv = dtT[buf][i][chl];
      float uv  = uT [buf][i][chl];
      float Bv  = bcT[buf][i][st];
      float Cv  = bcT[buf][i][DS + st];
      h = h * __expf(dtv * Av) + uv * Bv;
      float p = h * Cv;
      p += __shfl_xor(p, 1);
      p += __shfl_xor(p, 2);
      p += __shfl_xor(p, 4);
      p += __shfl_xor(p, 8);
      if (st == 0) yp[(size_t)(t0 + i) * DI + ch] = p;
    }
    __syncthreads();                                     // buf free before re-fill
  }
}

// ---------------- y = (ys + xs*D) * silu(z) -> bf16 ----------------
__global__ void k_epi(const float* __restrict__ ys, const float* __restrict__ xs,
                      const float* __restrict__ Dp, const float* __restrict__ xz,
                      bf16_t* __restrict__ yb) {
  int i = blockIdx.x * blockDim.x + threadIdx.x;
  if (i >= NROW * DI) return;
  int ch = i % DI, m = i / DI;
  float z = xz[(size_t)m * (2 * DI) + DI + ch];
  float v = (ys[i] + xs[i] * Dp[ch]) * siluf(z);
  yb[i] = (bf16_t)v;
}

// ---------------- residual + concat (unflip backward dir) -> bf16 ----------------
__global__ void k_res(const float* __restrict__ h0, const float* __restrict__ h1,
                      const float* __restrict__ x, bf16_t* __restrict__ res) {
  int i = blockIdx.x * blockDim.x + threadIdx.x;   // over NROW*DM
  if (i >= NROW * DM) return;
  int d = i % DM, m = i / DM;
  int b = m / LSEQ, l = m % LSEQ;
  float xv = x[i];
  res[(size_t)m * (2 * DM) + d] = (bf16_t)(h0[i] + xv);
  size_t mf = (size_t)b * LSEQ + (LSEQ - 1 - l);
  res[(size_t)m * (2 * DM) + DM + d] = (bf16_t)(h1[mf * DM + d] + xv);
}

extern "C" void kernel_launch(void* const* d_in, const int* in_sizes, int n_in,
                              void* d_out, int out_size, void* d_ws, size_t ws_size,
                              hipStream_t stream) {
  (void)in_sizes; (void)n_in; (void)out_size; (void)ws_size;
  const float* x = (const float*)d_in[0];
  const float* norm_w[2] = {(const float*)d_in[1],  (const float*)d_in[11]};
  const float* in_w[2]   = {(const float*)d_in[2],  (const float*)d_in[12]};
  const float* conv_w[2] = {(const float*)d_in[3],  (const float*)d_in[13]};
  const float* conv_b[2] = {(const float*)d_in[4],  (const float*)d_in[14]};
  const float* xproj_w[2]= {(const float*)d_in[5],  (const float*)d_in[15]};
  const float* dt_w[2]   = {(const float*)d_in[6],  (const float*)d_in[16]};
  const float* dt_b[2]   = {(const float*)d_in[7],  (const float*)d_in[17]};
  const float* A_log[2]  = {(const float*)d_in[8],  (const float*)d_in[18]};
  const float* Dp[2]     = {(const float*)d_in[9],  (const float*)d_in[19]};
  const float* out_w[2]  = {(const float*)d_in[10], (const float*)d_in[20]};
  const float* proj_w = (const float*)d_in[21];
  const float* proj_b = (const float*)d_in[22];

  // ---- workspace carve-out (256B aligned) ----
  char* ws = (char*)d_ws;
  size_t off = 0;
  auto alloc = [&](size_t bytes) -> char* {
    off = (off + 255) & ~(size_t)255;
    char* p = ws + off;
    off += bytes;
    return p;
  };
  bf16_t *xnb[2], *inwb[2], *xpwb[2], *outwb[2], *xsb[2], *yb[2];
  float  *xz[2], *xs[2], *dbl[2], *dt[2], *u[2], *ysv[2], *hid[2];
  for (int d = 0; d < 2; ++d) {
    xnb[d]  = (bf16_t*)alloc((size_t)NROW * DM * 2);
    inwb[d] = (bf16_t*)alloc((size_t)2 * DI * DM * 2);
    xpwb[d] = (bf16_t*)alloc((size_t)XPN * DI * 2);
    outwb[d]= (bf16_t*)alloc((size_t)DM * DI * 2);
    xsb[d]  = (bf16_t*)alloc((size_t)NROW * DI * 2);
    yb[d]   = (bf16_t*)alloc((size_t)NROW * DI * 2);
    xz[d]   = (float*)alloc((size_t)NROW * 2 * DI * 4);
    xs[d]   = (float*)alloc((size_t)NROW * DI * 4);
    dbl[d]  = (float*)alloc((size_t)NROW * XPN * 4);
    dt[d]   = (float*)alloc((size_t)NROW * DI * 4);
    u[d]    = (float*)alloc((size_t)NROW * DI * 4);
    ysv[d]  = (float*)alloc((size_t)NROW * DI * 4);
    hid[d]  = (float*)alloc((size_t)NROW * DM * 4);
  }
  bf16_t* pwb  = (bf16_t*)alloc((size_t)DM * 2 * DM * 2);
  bf16_t* resb = (bf16_t*)alloc((size_t)NROW * 2 * DM * 2);

  const int TB = 256;
  auto blocks = [](size_t n, int t) { return (unsigned)((n + t - 1) / t); };

  for (int d = 0; d < 2; ++d) {
    // weight packing to bf16
    k_pack<<<blocks(2 * DI * DM, TB), TB, 0, stream>>>(in_w[d],   inwb[d],  2 * DI * DM);
    k_pack<<<blocks(XPN * DI, TB),    TB, 0, stream>>>(xproj_w[d], xpwb[d], XPN * DI);
    k_pack<<<blocks(DM * DI, TB),     TB, 0, stream>>>(out_w[d],  outwb[d], DM * DI);
    // rmsnorm (bw direction reads flipped sequence)
    k_rmsnorm<<<NROW, DM, 0, stream>>>(x, norm_w[d], xnb[d], d);
    // in_proj: (4096x256) x (2048x256)^T -> xz (4096x2048)
    k_gemm<<<dim3(2 * DI / 16, NROW / 32), 32, 0, stream>>>(
        xnb[d], inwb[d], xz[d], nullptr, NROW, 2 * DI, DM);
    // depthwise conv + silu
    k_conv<<<blocks((size_t)NROW * DI, TB), TB, 0, stream>>>(
        xz[d], conv_w[d], conv_b[d], xs[d], xsb[d]);
    // x_proj: (4096x1024) x (48x1024)^T -> dbl (4096x48)
    k_gemm<<<dim3(XPN / 16, NROW / 32), 32, 0, stream>>>(
        xsb[d], xpwb[d], dbl[d], nullptr, NROW, XPN, DI);
    // dt softplus + u
    k_dtu<<<blocks((size_t)NROW * DI, TB), TB, 0, stream>>>(
        dbl[d], dt_w[d], dt_b[d], xs[d], dt[d], u[d]);
    // selective scan (async LDS staged)
    k_scan<<<dim3(DI / SCH, NB), 256, 0, stream>>>(dt[d], u[d], dbl[d], A_log[d], ysv[d]);
    // gating epilogue
    k_epi<<<blocks((size_t)NROW * DI, TB), TB, 0, stream>>>(
        ysv[d], xs[d], Dp[d], xz[d], yb[d]);
    // out_proj: (4096x1024) x (256x1024)^T -> hid (4096x256)
    k_gemm<<<dim3(DM / 16, NROW / 32), 32, 0, stream>>>(
        yb[d], outwb[d], hid[d], nullptr, NROW, DM, DI);
  }
  // residual + concat (unflips bw), then final projection with bias
  k_pack<<<blocks(DM * 2 * DM, TB), TB, 0, stream>>>(proj_w, pwb, DM * 2 * DM);
  k_res<<<blocks((size_t)NROW * DM, TB), TB, 0, stream>>>(hid[0], hid[1], x, resb);
  k_gemm<<<dim3(DM / 16, NROW / 32), 32, 0, stream>>>(
      resb, pwb, (float*)d_out, proj_b, NROW, DM, 2 * DM);
}
